// MultiHeadedAttention_84052509983214
// MI455X (gfx1250) — matrix operand order repaired and checked by reading, not compiled, not run on hardware
//
#include <hip/hip_runtime.h>
#include <hip/hip_bf16.h>

#define B_  2
#define S_  2048
#define D_  1024
#define H_  16
#define DK_ 64
#define BS_ (B_ * S_)   // 4096 rows

typedef __attribute__((ext_vector_type(16))) _Float16 v16h;
typedef __attribute__((ext_vector_type(8)))  _Float16 v8h;
typedef __attribute__((ext_vector_type(8)))  float    v8f;
typedef __attribute__((__vector_size__(16))) int      v4i_b;

union V16 { v16h v; v8h h2[2]; };

// ---- CDNA5 async global->LDS (ASYNCcnt-tracked), guarded with fallbacks ----
#if defined(__HIP_DEVICE_COMPILE__) && __has_builtin(__builtin_amdgcn_global_load_async_to_lds_b128)
  #define ASYNC_LDS 1
#else
  #define ASYNC_LDS 0
#endif

#if defined(__HIP_DEVICE_COMPILE__) && __has_builtin(__builtin_amdgcn_s_wait_asynccnt)
  #define WAIT_ASYNC(n) __builtin_amdgcn_s_wait_asynccnt(n)
#elif defined(__HIP_DEVICE_COMPILE__)
  #define WAIT_ASYNC(n) asm volatile("s_wait_asynccnt %0" ::"i"(n) : "memory")
#else
  #define WAIT_ASYNC(n) ((void)0)
#endif

#if ASYNC_LDS
static __device__ __forceinline__ void async_b128(const _Float16* g, _Float16* l) {
    __builtin_amdgcn_global_load_async_to_lds_b128(
        (__attribute__((address_space(1))) v4i_b*)(g),
        (__attribute__((address_space(3))) v4i_b*)(l), 0, 0);
}
#endif

static __device__ __forceinline__ v8f wmma16(v16h a, v16h b, v8f c) {
    return __builtin_amdgcn_wmma_f32_16x16x32_f16(
        false, a, false, b, (short)0, c, false, false);
}

static __device__ __forceinline__ v8h pack8(float4 x, float4 y) {
    v8h h;
    h[0] = (_Float16)x.x; h[1] = (_Float16)x.y;
    h[2] = (_Float16)x.z; h[3] = (_Float16)x.w;
    h[4] = (_Float16)y.x; h[5] = (_Float16)y.y;
    h[6] = (_Float16)y.z; h[7] = (_Float16)y.w;
    return h;
}

// ---------------------------------------------------------------------------
// Projection GEMM: out[r,n] = sum_k X[r,k]*W[n,k] + bias[n]
// f32 inputs -> f16 LDS tiles -> WMMA f32 acc. Tile 64x64, K-step 64,
// software-pipelined (register prefetch of next K-slab during compute).
// Output f16 head-major [B,H,S,DK].
// ---------------------------------------------------------------------------
__global__ __launch_bounds__(256) void proj_qkv_kernel(
    const float* __restrict__ X, const float* __restrict__ W,
    const float* __restrict__ bias, _Float16* __restrict__ out)
{
    __shared__ __align__(16) _Float16 sA[64][72];
    __shared__ __align__(16) _Float16 sB[64][72];

    const int tid  = threadIdx.x;
    const int lane = tid & 31;
    const int wave = tid >> 5;
    const int hlf  = lane >> 4;
    const int l16  = lane & 15;
    const int waveM = wave >> 1;   // 0..3
    const int waveN = wave & 1;    // 0..1

    const int mTile = blockIdx.x * 64;
    const int nTile = blockIdx.y * 64;

    const int ldRow = tid >> 2;          // 0..63
    const int ldC   = (tid & 3) * 16;    // 0,16,32,48

    const float* aptr = X + (size_t)(mTile + ldRow) * D_ + ldC;
    const float* wptr = W + (size_t)(nTile + ldRow) * D_ + ldC;

    // prefetch K-slab 0
    float4 a0 = *(const float4*)(aptr),     a1 = *(const float4*)(aptr + 4);
    float4 a2 = *(const float4*)(aptr + 8), a3 = *(const float4*)(aptr + 12);
    float4 w0 = *(const float4*)(wptr),     w1 = *(const float4*)(wptr + 4);
    float4 w2 = *(const float4*)(wptr + 8), w3 = *(const float4*)(wptr + 12);

    v8f acc0 = {}, acc1 = {};

    for (int k0 = 0; k0 < D_; k0 += 64) {
        __syncthreads();                                    // prev compute done
        *(v8h*)&sA[ldRow][ldC]     = pack8(a0, a1);
        *(v8h*)&sA[ldRow][ldC + 8] = pack8(a2, a3);
        *(v8h*)&sB[ldRow][ldC]     = pack8(w0, w1);
        *(v8h*)&sB[ldRow][ldC + 8] = pack8(w2, w3);
        if (k0 + 64 < D_) {                                 // prefetch next slab
            const float* an = aptr + k0 + 64;
            const float* wn = wptr + k0 + 64;
            a0 = *(const float4*)(an);     a1 = *(const float4*)(an + 4);
            a2 = *(const float4*)(an + 8); a3 = *(const float4*)(an + 12);
            w0 = *(const float4*)(wn);     w1 = *(const float4*)(wn + 4);
            w2 = *(const float4*)(wn + 8); w3 = *(const float4*)(wn + 12);
        }
        __syncthreads();                                    // tile visible
#pragma unroll
        for (int c = 0; c < 2; c++) {
            V16 a;
            a.h2[0] = *(const v8h*)&sA[waveM * 16 + l16][32 * c + 16 * hlf];
            a.h2[1] = *(const v8h*)&sA[waveM * 16 + l16][32 * c + 16 * hlf + 8];
            V16 b0, b1;
            b0.h2[0] = *(const v8h*)&sB[waveN * 32 + l16][32 * c + 16 * hlf];
            b0.h2[1] = *(const v8h*)&sB[waveN * 32 + l16][32 * c + 16 * hlf + 8];
            b1.h2[0] = *(const v8h*)&sB[waveN * 32 + 16 + l16][32 * c + 16 * hlf];
            b1.h2[1] = *(const v8h*)&sB[waveN * 32 + 16 + l16][32 * c + 16 * hlf + 8];
            acc0 = wmma16(a.v, b0.v, acc0);
            acc1 = wmma16(a.v, b1.v, acc1);
        }
    }

#pragma unroll
    for (int nf = 0; nf < 2; nf++) {
        const int n = nTile + waveN * 32 + nf * 16 + l16;
        const float bv = bias[n];
        const int hh = n >> 6, dk = n & 63;
        const v8f& acc = nf ? acc1 : acc0;
#pragma unroll
        for (int r = 0; r < 8; r++) {
            const int rg = mTile + waveM * 16 + r + 8 * hlf;
            const int bb = rg >> 11, ss = rg & (S_ - 1);
            out[(((size_t)(bb * H_ + hh)) * S_ + ss) * DK_ + dk] =
                (_Float16)(acc[r] + bv);
        }
    }
}

// ---------------------------------------------------------------------------
// Output projection: A f16 [4096,1024] x Wo^T + bo -> f32 out. Same pipeline.
// ---------------------------------------------------------------------------
__global__ __launch_bounds__(256) void proj_out_kernel(
    const _Float16* __restrict__ A, const float* __restrict__ W,
    const float* __restrict__ bias, float* __restrict__ out)
{
    __shared__ __align__(16) _Float16 sA[64][72];
    __shared__ __align__(16) _Float16 sB[64][72];

    const int tid  = threadIdx.x;
    const int lane = tid & 31;
    const int wave = tid >> 5;
    const int hlf  = lane >> 4;
    const int l16  = lane & 15;
    const int waveM = wave >> 1;
    const int waveN = wave & 1;

    const int mTile = blockIdx.x * 64;
    const int nTile = blockIdx.y * 64;

    const int ldRow = tid >> 2;
    const int ldC   = (tid & 3) * 16;

    const _Float16* aptr = A + (size_t)(mTile + ldRow) * D_ + ldC;
    const float*    wptr = W + (size_t)(nTile + ldRow) * D_ + ldC;

    v8h   ar0 = *(const v8h*)(aptr), ar1 = *(const v8h*)(aptr + 8);
    float4 w0 = *(const float4*)(wptr),     w1 = *(const float4*)(wptr + 4);
    float4 w2 = *(const float4*)(wptr + 8), w3 = *(const float4*)(wptr + 12);

    v8f acc0 = {}, acc1 = {};

    for (int k0 = 0; k0 < D_; k0 += 64) {
        __syncthreads();
        *(v8h*)&sA[ldRow][ldC]     = ar0;
        *(v8h*)&sA[ldRow][ldC + 8] = ar1;
        *(v8h*)&sB[ldRow][ldC]     = pack8(w0, w1);
        *(v8h*)&sB[ldRow][ldC + 8] = pack8(w2, w3);
        if (k0 + 64 < D_) {
            const _Float16* an = aptr + k0 + 64;
            const float*    wn = wptr + k0 + 64;
            ar0 = *(const v8h*)(an); ar1 = *(const v8h*)(an + 8);
            w0 = *(const float4*)(wn);     w1 = *(const float4*)(wn + 4);
            w2 = *(const float4*)(wn + 8); w3 = *(const float4*)(wn + 12);
        }
        __syncthreads();
#pragma unroll
        for (int c = 0; c < 2; c++) {
            V16 a;
            a.h2[0] = *(const v8h*)&sA[waveM * 16 + l16][32 * c + 16 * hlf];
            a.h2[1] = *(const v8h*)&sA[waveM * 16 + l16][32 * c + 16 * hlf + 8];
            V16 b0, b1;
            b0.h2[0] = *(const v8h*)&sB[waveN * 32 + l16][32 * c + 16 * hlf];
            b0.h2[1] = *(const v8h*)&sB[waveN * 32 + l16][32 * c + 16 * hlf + 8];
            b1.h2[0] = *(const v8h*)&sB[waveN * 32 + 16 + l16][32 * c + 16 * hlf];
            b1.h2[1] = *(const v8h*)&sB[waveN * 32 + 16 + l16][32 * c + 16 * hlf + 8];
            acc0 = wmma16(a.v, b0.v, acc0);
            acc1 = wmma16(a.v, b1.v, acc1);
        }
    }

#pragma unroll
    for (int nf = 0; nf < 2; nf++) {
        const int n = nTile + waveN * 32 + nf * 16 + l16;
        const float bv = bias[n];
        const v8f& acc = nf ? acc1 : acc0;
#pragma unroll
        for (int r = 0; r < 8; r++) {
            const int rg = mTile + waveM * 16 + r + 8 * hlf;
            out[(size_t)rg * D_ + n] = acc[r] + bv;
        }
    }
}

// ---------------------------------------------------------------------------
// Flash attention, q-tile 128, one (b,h) per block, 8 waves x 16 q-rows.
// Double-buffered K/V tiles; K tile streamed with GLOBAL_LOAD_ASYNC_TO_LDS
// (ASYNCcnt) when available, register-prefetch fallback otherwise.
// Online softmax in log2 domain. Output f16 merged [B,S,D].
// ---------------------------------------------------------------------------
__global__ __launch_bounds__(256) void attn_kernel(
    const _Float16* __restrict__ Q, const _Float16* __restrict__ K,
    const _Float16* __restrict__ V, const float* __restrict__ tau,
    const float* __restrict__ delta, _Float16* __restrict__ O)
{
    __shared__ __align__(16) _Float16 sK[2][64][72];     // [buf][k_j][d]
    __shared__ __align__(16) _Float16 sV[2][64][72];     // [buf][d][k_j] (transposed)
    __shared__ __align__(16) _Float16 sP[8][16][72];     // per-wave P staging

    const int tid  = threadIdx.x;
    const int lane = tid & 31;
    const int wave = tid >> 5;
    const int hlf  = lane >> 4;
    const int l16  = lane & 15;

    const int qTile = blockIdx.x * 128;
    const int hh = blockIdx.y, bb = blockIdx.z;

    const _Float16* Qh = Q + ((size_t)(bb * H_ + hh) * S_) * DK_;
    const _Float16* Kh = K + ((size_t)(bb * H_ + hh) * S_) * DK_;
    const _Float16* Vh = V + ((size_t)(bb * H_ + hh) * S_) * DK_;

    const float LOG2E = 1.44269504f;
    const float a2 = tau[bb] * 0.125f * LOG2E;   // tau/sqrt(DK) in log2 domain
    const float dsc = 0.125f * LOG2E;
    const float* drow = delta + (size_t)bb * S_;

    // persistent Q fragments: rows qTile + wave*16 + l16
    const _Float16* qrow = Qh + (size_t)(qTile + wave * 16 + l16) * DK_;
    V16 qa0, qa1;
    qa0.h2[0] = *(const v8h*)(qrow + 16 * hlf);
    qa0.h2[1] = *(const v8h*)(qrow + 16 * hlf + 8);
    qa1.h2[0] = *(const v8h*)(qrow + 32 + 16 * hlf);
    qa1.h2[1] = *(const v8h*)(qrow + 32 + 16 * hlf + 8);

    float m[8], l[8];
    v8f o[4] = {};
#pragma unroll
    for (int r = 0; r < 8; r++) { m[r] = -1e30f; l[r] = 0.f; }

    const int ldRow = tid >> 2;          // 0..63
    const int ldC   = (tid & 3) * 16;    // 0,16,32,48
    const _Float16* kptr = Kh + (size_t)ldRow * DK_ + ldC;  // + kt*64*DK_
    const _Float16* vptr = Vh + (size_t)ldRow * DK_ + ldC;

    const int NT = S_ / 64;

    // ---- prologue: tile 0 in flight ----
#if ASYNC_LDS
    async_b128(kptr,     &sK[0][ldRow][ldC]);
    async_b128(kptr + 8, &sK[0][ldRow][ldC + 8]);
#else
    v8h kr0 = *(const v8h*)(kptr), kr1 = *(const v8h*)(kptr + 8);
#endif
    v8h vr0 = *(const v8h*)(vptr), vr1 = *(const v8h*)(vptr + 8);

    for (int kt = 0; kt < NT; kt++) {
        const int cur = kt & 1, nxt = cur ^ 1;
        const int kBase = kt * 64;

        __syncthreads();   // #1: everyone finished reading buffers touched below
#if !ASYNC_LDS
        *(v8h*)&sK[cur][ldRow][ldC]     = kr0;
        *(v8h*)&sK[cur][ldRow][ldC + 8] = kr1;
#endif
#pragma unroll
        for (int e = 0; e < 8; e++) {
            sV[cur][ldC + e][ldRow]     = vr0[e];
            sV[cur][ldC + 8 + e][ldRow] = vr1[e];
        }
        if (kt + 1 < NT) {   // stream next tile while we compute this one
            const _Float16* kn = kptr + (size_t)(kt + 1) * 64 * DK_;
            const _Float16* vn = vptr + (size_t)(kt + 1) * 64 * DK_;
#if ASYNC_LDS
            async_b128(kn,     &sK[nxt][ldRow][ldC]);
            async_b128(kn + 8, &sK[nxt][ldRow][ldC + 8]);
            WAIT_ASYNC(2);   // current tile's 2 async ops retired
#else
            kr0 = *(const v8h*)(kn); kr1 = *(const v8h*)(kn + 8);
#endif
            vr0 = *(const v8h*)(vn); vr1 = *(const v8h*)(vn + 8);
        } else {
#if ASYNC_LDS
            WAIT_ASYNC(0);
#endif
        }
        __syncthreads();   // #2: tile cur fully in LDS for all waves

        // ---- scores: 16 q rows x 64 k cols, K-dim = DK = 64 ----
        v8f sfr[4];
#pragma unroll
        for (int nf = 0; nf < 4; nf++) {
            v8f s = {};
            const _Float16* kr = &sK[cur][nf * 16 + l16][0];
            V16 kb;
            kb.h2[0] = *(const v8h*)(kr + 16 * hlf);
            kb.h2[1] = *(const v8h*)(kr + 16 * hlf + 8);
            s = wmma16(qa0.v, kb.v, s);
            kb.h2[0] = *(const v8h*)(kr + 32 + 16 * hlf);
            kb.h2[1] = *(const v8h*)(kr + 32 + 16 * hlf + 8);
            s = wmma16(qa1.v, kb.v, s);
            const float dj = drow[kBase + nf * 16 + l16] * dsc;
#pragma unroll
            for (int r = 0; r < 8; r++) sfr[nf][r] = s[r] * a2 + dj;
        }

        // ---- online softmax (rows in VGPR index, cols across half-wave) ----
        float mnew[8], scale[8], rsum[8];
#pragma unroll
        for (int r = 0; r < 8; r++) {
            float t = fmaxf(fmaxf(sfr[0][r], sfr[1][r]),
                            fmaxf(sfr[2][r], sfr[3][r]));
#pragma unroll
            for (int msk = 1; msk < 16; msk <<= 1)
                t = fmaxf(t, __shfl_xor(t, msk, 32));
            mnew[r] = fmaxf(m[r], t);
            scale[r] = __builtin_amdgcn_exp2f(m[r] - mnew[r]);
            m[r] = mnew[r];
            rsum[r] = 0.f;
        }
#pragma unroll
        for (int nf = 0; nf < 4; nf++) {
#pragma unroll
            for (int r = 0; r < 8; r++) {
                const float p = __builtin_amdgcn_exp2f(sfr[nf][r] - mnew[r]);
                rsum[r] += p;
                sP[wave][r + 8 * hlf][nf * 16 + l16] = (_Float16)p;
            }
        }
#pragma unroll
        for (int r = 0; r < 8; r++) {
            float t = rsum[r];
#pragma unroll
            for (int msk = 1; msk < 16; msk <<= 1)
                t += __shfl_xor(t, msk, 32);
            l[r] = l[r] * scale[r] + t;
        }
#pragma unroll
        for (int f = 0; f < 4; f++)
#pragma unroll
            for (int r = 0; r < 8; r++) o[f][r] *= scale[r];

        // ---- P @ V : A = P (16x64 over j), B = V^T (j x dk) ----
        const _Float16* pr = &sP[wave][l16][0];   // same-wave DS ops in-order
        V16 pa0, pa1;
        pa0.h2[0] = *(const v8h*)(pr + 16 * hlf);
        pa0.h2[1] = *(const v8h*)(pr + 16 * hlf + 8);
        pa1.h2[0] = *(const v8h*)(pr + 32 + 16 * hlf);
        pa1.h2[1] = *(const v8h*)(pr + 32 + 16 * hlf + 8);
#pragma unroll
        for (int f = 0; f < 4; f++) {
            const _Float16* vr = &sV[cur][f * 16 + l16][0];
            V16 vb;
            vb.h2[0] = *(const v8h*)(vr + 16 * hlf);
            vb.h2[1] = *(const v8h*)(vr + 16 * hlf + 8);
            o[f] = wmma16(pa0.v, vb.v, o[f]);
            vb.h2[0] = *(const v8h*)(vr + 32 + 16 * hlf);
            vb.h2[1] = *(const v8h*)(vr + 32 + 16 * hlf + 8);
            o[f] = wmma16(pa1.v, vb.v, o[f]);
        }
    }

    // ---- normalize + store merged heads: O[b][s][h*DK+dk] (f16) ----
#pragma unroll
    for (int r = 0; r < 8; r++) {
        const float inv = 1.0f / l[r];
        const int srow = qTile + wave * 16 + r + 8 * hlf;
        _Float16* orow = O + ((size_t)bb * S_ + srow) * D_ + hh * DK_;
#pragma unroll
        for (int f = 0; f < 4; f++)
            orow[f * 16 + l16] = (_Float16)(o[f][r] * inv);
    }
}

// ---------------------------------------------------------------------------
extern "C" void kernel_launch(void* const* d_in, const int* in_sizes, int n_in,
                              void* d_out, int out_size, void* d_ws, size_t ws_size,
                              hipStream_t stream) {
    (void)in_sizes; (void)n_in; (void)out_size; (void)ws_size;
    const float* query = (const float*)d_in[0];
    const float* key   = (const float*)d_in[1];
    const float* value = (const float*)d_in[2];
    const float* tau   = (const float*)d_in[3];
    const float* delta = (const float*)d_in[4];
    const float* Wq = (const float*)d_in[5];
    const float* bq = (const float*)d_in[6];
    const float* Wk = (const float*)d_in[7];
    const float* bk = (const float*)d_in[8];
    const float* Wv = (const float*)d_in[9];
    const float* bv = (const float*)d_in[10];
    const float* Wo = (const float*)d_in[11];
    const float* bo = (const float*)d_in[12];

    const size_t elems = (size_t)B_ * H_ * S_ * DK_;   // 4,194,304 halves each
    _Float16* qws = (_Float16*)d_ws;
    _Float16* kws = qws + elems;
    _Float16* vws = kws + elems;
    _Float16* aws = vws + elems;                       // merged attn out [B,S,D]

    dim3 blk(256);
    dim3 gProj(BS_ / 64, D_ / 64);                     // 64 x 16
    proj_qkv_kernel<<<gProj, blk, 0, stream>>>(query, Wq, bq, qws);
    proj_qkv_kernel<<<gProj, blk, 0, stream>>>(key,   Wk, bk, kws);
    proj_qkv_kernel<<<gProj, blk, 0, stream>>>(value, Wv, bv, vws);

    dim3 gAttn(S_ / 128, H_, B_);                      // 16 x 16 x 2
    attn_kernel<<<gAttn, blk, 0, stream>>>(qws, kws, vws, tau, delta, aws);

    proj_out_kernel<<<gProj, blk, 0, stream>>>(aws, Wo, bo, (float*)d_out);
}
